// S2NetReconstruction_69965017252360
// MI455X (gfx1250) — matrix operand order repaired and checked by reading, not compiled, not run on hardware
//
#include <hip/hip_runtime.h>
#include <hip/hip_bf16.h>

typedef __attribute__((ext_vector_type(16))) _Float16 v16h;
typedef __attribute__((ext_vector_type(8)))  float    v8f;

#define B_   128
#define T_   512
#define N_   116
#define D_   4
#define H_   256
#define N2_  232
#define NP   128   // N padded to WMMA K multiple
#define N2P  256   // 2N padded
#define DT_   0.1f
#define KC_   1.0f
#define BETA_ 0.9f
#define THR_  1.0f

// ---------------------------------------------------------------- WMMA helpers

__device__ __forceinline__ v8f wmma16(v16h a, v16h b, v8f c) {
  // D = A(16x32 f16) * B(32x16 f16) + C(16x16 f32)
  return __builtin_amdgcn_wmma_f32_16x16x32_f16(false, a, false, b, (short)0, c,
                                                false, false);
}

// A-matrix fragment from row-major f16 LDS tile (lda in halfwords).
// Layout used here (contiguous K, not AMD's table order): lane rows M=lane%16, K = kb..kb+15 contiguous
// with kb = 0 (lanes 0-15) or 16 (lanes 16-31).
__device__ __forceinline__ v16h load_A(const _Float16* p, int row_base, int k_base, int lda) {
  const int lane = threadIdx.x & 31;
  const _Float16* r = p + (size_t)(row_base + (lane & 15)) * lda + k_base + ((lane < 16) ? 0 : 16);
  v16h a;
#pragma unroll
  for (int i = 0; i < 16; ++i) a[i] = r[i];
  return a;
}

// B-matrix fragment from pre-swizzled global storage: per (tile,lane) 16 contiguous halfwords.
__device__ __forceinline__ v16h load_Bfrag(const _Float16* frag, int tile) {
  const int lane = threadIdx.x & 31;
  return *(const v16h*)(frag + ((size_t)tile * 32 + lane) * 16);
}

// B-matrix fragment from column-major f16 LDS (BmT[col][ldk]): lane col = lane%16,
// K = kb..kb+15 contiguous -> two ds_load_b128 per fragment.
__device__ __forceinline__ v16h load_BldsT(const _Float16* p, int k_base, int ldk) {
  const int lane = threadIdx.x & 31;
  const _Float16* q = p + (size_t)(lane & 15) * ldk + k_base + ((lane < 16) ? 0 : 16);
  v16h b;
#pragma unroll
  for (int i = 0; i < 16; ++i) b[i] = q[i];
  return b;
}

// ------------------------------------------------- weight packing (once, cheap)

__global__ void pack_weight_frags(const float* __restrict__ W, _Float16* __restrict__ frag,
                                  int K, int Ncols, int KT, int NT) {
  const int total = KT * NT * 32;
  for (int f = blockIdx.x * blockDim.x + threadIdx.x; f < total; f += gridDim.x * blockDim.x) {
    const int lane = f & 31, tile = f >> 5;
    const int nt = tile % NT, kt = tile / NT;
    const int n  = nt * 16 + (lane & 15);
    const int kb = kt * 32 + ((lane < 16) ? 0 : 16);
    _Float16* dst = frag + (size_t)tile * 512 + (size_t)lane * 16;
#pragma unroll
    for (int h = 0; h < 16; ++h) {
      const int k = kb + h;
      const float v = (k < K && n < Ncols) ? W[(size_t)k * Ncols + n] : 0.f;
      dst[h] = (_Float16)v;
    }
  }
}

// ----------------------------------------------------------- fused encoder
// gamma = leaky(LN(x@Wf+bf))@We + be, one 16-token tile per workgroup (4 waves).

__global__ __launch_bounds__(128)
void encoder_kernel(const float* __restrict__ x, const _Float16* __restrict__ WfF,
                    const float* __restrict__ bf, const float* __restrict__ gf,
                    const float* __restrict__ betaf, const _Float16* __restrict__ WeF,
                    const float* __restrict__ be, float* __restrict__ gamma) {
  __shared__ _Float16 xA[16 * NP];
  __shared__ float    zf[16 * H_];
  __shared__ _Float16 zA[16 * H_];
  const int tid = threadIdx.x, wave = tid >> 5, lane = tid & 31;
  const int row0 = blockIdx.x * 16;

  for (int idx = tid; idx < 16 * NP; idx += 128) {
    const int r = idx >> 7, c = idx & (NP - 1);
    xA[idx] = (_Float16)((c < N_) ? x[(size_t)(row0 + r) * N_ + c] : 0.f);
  }
  __syncthreads();

  // GEMM1: [16,128] @ Wf[128,256]
  for (int nt = wave * 4; nt < wave * 4 + 4; ++nt) {
    v8f acc = {};
#pragma unroll
    for (int kt = 0; kt < 4; ++kt) {
      if (kt < 3) __builtin_prefetch(WfF + ((size_t)((kt + 1) * 16 + nt) * 32 + lane) * 16, 0, 1);
      acc = wmma16(load_A(xA, 0, kt * 32, NP), load_Bfrag(WfF, kt * 16 + nt), acc);
    }
    const int col = nt * 16 + (lane & 15);
    const int rb  = (lane < 16) ? 0 : 8;
    const float bias = bf[col];
#pragma unroll
    for (int r = 0; r < 8; ++r) zf[(rb + r) * H_ + col] = acc[r] + bias;
  }
  __syncthreads();

  if (tid < 16) {  // LayerNorm + leaky over H=256
    float m = 0.f, m2 = 0.f;
    for (int c = 0; c < H_; ++c) { const float v = zf[tid * H_ + c]; m += v; m2 += v * v; }
    m *= (1.f / H_); m2 = m2 * (1.f / H_) - m * m;
    const float inv = rsqrtf(m2 + 1e-5f);
    for (int c = 0; c < H_; ++c) {
      float v = (zf[tid * H_ + c] - m) * inv * gf[c] + betaf[c];
      v = (v > 0.f) ? v : 0.1f * v;
      zA[tid * H_ + c] = (_Float16)v;
    }
  }
  __syncthreads();

  // GEMM2: [16,256] @ We[256,128] -> gamma (cols < 116)
  for (int nt = wave * 2; nt < wave * 2 + 2; ++nt) {
    v8f acc = {};
#pragma unroll
    for (int kt = 0; kt < 8; ++kt)
      acc = wmma16(load_A(zA, 0, kt * 32, H_), load_Bfrag(WeF, kt * 8 + nt), acc);
    const int col = nt * 16 + (lane & 15);
    const int rb  = (lane < 16) ? 0 : 8;
    if (col < N_) {
      const float bias = be[col];
#pragma unroll
      for (int r = 0; r < 8; ++r)
        gamma[(size_t)(row0 + rb + r) * N_ + col] = acc[r] + bias;
    }
  }
}

// ----------------------------------------------------------- Kuramoto scan
// One workgroup per batch element. sc[b] lives in LDS (f16, A-layout source),
// theta/sin/cos live in registers (thread i owns region i). Per step:
//   [As|Ac] = sc @ [sin|cos]  via 32x v_wmma_f32_16x16x32_f16,
//   theta += DT*(omega + gamma_t + KC/N * (c*As - s*Ac)),
// emitting mask[t] (delayed by 2) and u[t] = sum_d sin(theta)*w_snn[d].
// gamma rows are double-buffered in LDS via global_load_async_to_lds_b32
// (ASYNCcnt), so the HBM/L2 fetch of row t+1 overlaps step t's sincos and step t+1's WMMA.

__global__ __launch_bounds__(128)
void kuramoto_kernel(const float* __restrict__ sc, const float* __restrict__ gamma,
                     const float* __restrict__ omega, const float* __restrict__ w_snn,
                     float* __restrict__ mask_out, float* __restrict__ u_out) {
  __shared__ _Float16 scA[NP * NP];    // 32 KB, A-layout source
  __shared__ _Float16 BmT[16 * NP];    //  4 KB, column-major: BmT[col][j]
  __shared__ float    AsAc[NP * 8];    //  4 KB
  __shared__ float    gbuf[2 * NP];    //  1 KB, gamma row double buffer
  const int tid = threadIdx.x, wave = tid >> 5, lane = tid & 31;
  const int b = blockIdx.x;

  for (int idx = tid; idx < NP * NP; idx += 128) {
    const int r = idx >> 7, c = idx & (NP - 1);
    scA[idx] = (_Float16)((r < N_ && c < N_) ? sc[((size_t)b * N_ + r) * N_ + c] : 0.f);
  }
  for (int idx = tid; idx < 16 * NP; idx += 128) BmT[idx] = (_Float16)0.f;

  float th[D_] = {0.f, 0.f, 0.f, 0.f};
  float s[D_]  = {0.f, 0.f, 0.f, 0.f};
  float c[D_]  = {1.f, 1.f, 1.f, 1.f};
  float om[D_] = {0.f, 0.f, 0.f, 0.f};
  float w[D_];
#pragma unroll
  for (int d = 0; d < D_; ++d) w[d] = w_snn[d];
  if (tid < N_) {
#pragma unroll
    for (int d = 0; d < D_; ++d) {
      om[d] = omega[tid * D_ + d];
      BmT[d * NP + tid]       = (_Float16)0.f;  // sin(0)
      BmT[(4 + d) * NP + tid] = (_Float16)1.f;  // cos(0)
    }
    // async prefetch of gamma row t=0 into buffer 0 (ASYNCcnt path)
    {
      const unsigned dst = (unsigned)(uintptr_t)(const void*)&gbuf[tid];
      const unsigned long long src =
          (unsigned long long)(const void*)&gamma[((size_t)b * T_ + 0) * N_ + tid];
      asm volatile("global_load_async_to_lds_b32 %0, %1, off"
                   :: "v"(dst), "v"(src) : "memory");
    }
  }
  float d1 = 0.f, d2 = 0.f, bar0 = 0.f;
  __syncthreads();

  for (int t = 0; t < T_; ++t) {
    // ---- coupling GEMM: [As|Ac] = sc @ [sin|cos] (2 M-tiles per wave)
    for (int m = wave * 2; m < wave * 2 + 2; ++m) {
      v8f acc = {};
#pragma unroll
      for (int kt = 0; kt < 4; ++kt)
        acc = wmma16(load_A(scA, m * 16, kt * 32, NP), load_BldsT(BmT, kt * 32, NP), acc);
      const int col = lane & 15;
      if (col < 8) {
        const int rb = m * 16 + ((lane < 16) ? 0 : 8);
#pragma unroll
        for (int r = 0; r < 8; ++r) AsAc[(rb + r) * 8 + col] = acc[r];
      }
    }
    __syncthreads();

    // ---- per-region phase update + streaming mask/u emission
    if (tid < N_) {
      asm volatile("s_wait_asynccnt 0x0" ::: "memory");   // gamma row t is in LDS
      const float g = gbuf[(t & 1) * NP + tid];
      if (t + 1 < T_) {  // overlap next row's fetch with this step's math
        const unsigned dst = (unsigned)(uintptr_t)(const void*)&gbuf[((t + 1) & 1) * NP + tid];
        const unsigned long long src =
            (unsigned long long)(const void*)&gamma[((size_t)b * T_ + (t + 1)) * N_ + tid];
        asm volatile("global_load_async_to_lds_b32 %0, %1, off"
                     :: "v"(dst), "v"(src) : "memory");
      }
      float bar = 0.f, u = 0.f;
#pragma unroll
      for (int d = 0; d < D_; ++d) {
        const float coup = c[d] * AsAc[tid * 8 + d] - s[d] * AsAc[tid * 8 + 4 + d];
        th[d] += DT_ * (om[d] + g + (KC_ / (float)N_) * coup);
        bar += th[d];
        s[d] = __sinf(th[d]);
        c[d] = __cosf(th[d]);
        u += s[d] * w[d];
        BmT[d * NP + tid]       = (_Float16)s[d];   // operand for next step
        BmT[(4 + d) * NP + tid] = (_Float16)c[d];
      }
      bar *= (1.f / (float)D_);
      if (t == 0) bar0 = bar;
      const float mb = (t >= 2) ? d2 : bar0;        // thetas[max(t-2,0)].mean(-1)
      d2 = d1; d1 = bar;
      const float mv = 0.5f * (1.f + __sinf(mb));
      const size_t o = ((size_t)t * B_ + b) * N_ + tid;
      mask_out[o] = mv;
      u_out[o]    = u;
    }
    __syncthreads();
  }
}

// ----------------------------------------------------------- gating GEMM
// I = (mask*u + b_snn) * sigmoid(mask@Wm + bm), rows in [T*B)

__global__ __launch_bounds__(128)
void gating_kernel(const float* __restrict__ mask, const float* __restrict__ u,
                   const _Float16* __restrict__ WmF, const float* __restrict__ bm,
                   const float* __restrict__ b_snn, float* __restrict__ Iout) {
  __shared__ _Float16 mA[16 * NP];
  const int tid = threadIdx.x, wave = tid >> 5, lane = tid & 31;
  const int row0 = blockIdx.x * 16;

  for (int idx = tid; idx < 16 * NP; idx += 128) {
    const int r = idx >> 7, cc = idx & (NP - 1);
    mA[idx] = (_Float16)((cc < N_) ? mask[(size_t)(row0 + r) * N_ + cc] : 0.f);
  }
  __syncthreads();

  for (int nt = wave * 2; nt < wave * 2 + 2; ++nt) {
    v8f acc = {};
#pragma unroll
    for (int kt = 0; kt < 4; ++kt)
      acc = wmma16(load_A(mA, 0, kt * 32, NP), load_Bfrag(WmF, kt * 8 + nt), acc);
    const int col = nt * 16 + (lane & 15);
    const int rb  = (lane < 16) ? 0 : 8;
    if (col < N_) {
      const float bias = bm[col], bs = b_snn[col];
#pragma unroll
      for (int r = 0; r < 8; ++r) {
        const size_t o = (size_t)(row0 + rb + r) * N_ + col;
        const float gt = 1.f / (1.f + __expf(-(acc[r] + bias)));
        Iout[o] = (mask[o] * u[o] + bs) * gt;
      }
    }
  }
}

// ----------------------------------------------------------- LIF recurrence

__global__ void lif_kernel(const float* __restrict__ I, float* __restrict__ snn,
                           float* __restrict__ spikes) {
  const int e = blockIdx.x * blockDim.x + threadIdx.x;
  if (e >= B_ * N_) return;
  const int b = e / N_, n = e % N_;
  float v = 0.f, sp = 0.f;
  for (int t = 0; t < T_; ++t) {
    const float It = I[((size_t)t * B_ + b) * N_ + n];
    v = BETA_ * v * (1.f - sp) + It;
    sp = (v > THR_) ? 1.f : 0.f;  // forward value of surrogate trick == hard spike
    const size_t o = ((size_t)b * T_ + t) * N_ + n;
    snn[o]    = v;
    spikes[o] = sp;
  }
}

// ----------------------------------------------------------- fused decoder
// recon = leaky(LN(leaky(LN(snn@Wd1+bd1))@Wd2+bd2))@Wd3 + bd3

__global__ __launch_bounds__(128)
void decoder_kernel(const float* __restrict__ snn,
                    const _Float16* __restrict__ Wd1F, const float* __restrict__ bd1,
                    const float* __restrict__ g1, const float* __restrict__ be1,
                    const _Float16* __restrict__ Wd2F, const float* __restrict__ bd2,
                    const float* __restrict__ g2, const float* __restrict__ be2,
                    const _Float16* __restrict__ Wd3F, const float* __restrict__ bd3,
                    float* __restrict__ recon) {
  __shared__ _Float16 inA[16 * NP];
  __shared__ float    hf[16 * N2P];
  __shared__ _Float16 aA[16 * N2P];
  const int tid = threadIdx.x, wave = tid >> 5, lane = tid & 31;
  const int row0 = blockIdx.x * 16;

  for (int idx = tid; idx < 16 * NP; idx += 128) {
    const int r = idx >> 7, cc = idx & (NP - 1);
    inA[idx] = (_Float16)((cc < N_) ? snn[(size_t)(row0 + r) * N_ + cc] : 0.f);
  }
  __syncthreads();

  // GEMM1: [16,128] @ Wd1[128,256]
  for (int nt = wave * 4; nt < wave * 4 + 4; ++nt) {
    v8f acc = {};
#pragma unroll
    for (int kt = 0; kt < 4; ++kt)
      acc = wmma16(load_A(inA, 0, kt * 32, NP), load_Bfrag(Wd1F, kt * 16 + nt), acc);
    const int col = nt * 16 + (lane & 15);
    const int rb  = (lane < 16) ? 0 : 8;
    const float bias = (col < N2_) ? bd1[col] : 0.f;
#pragma unroll
    for (int r = 0; r < 8; ++r) hf[(rb + r) * N2P + col] = acc[r] + bias;
  }
  __syncthreads();

  if (tid < 16) {  // LN1 + leaky over 232 valid cols
    float m = 0.f, m2 = 0.f;
    for (int cc = 0; cc < N2_; ++cc) { const float v = hf[tid * N2P + cc]; m += v; m2 += v * v; }
    m *= (1.f / N2_); m2 = m2 * (1.f / N2_) - m * m;
    const float inv = rsqrtf(m2 + 1e-5f);
    for (int cc = 0; cc < N2P; ++cc) {
      float v = 0.f;
      if (cc < N2_) {
        v = (hf[tid * N2P + cc] - m) * inv * g1[cc] + be1[cc];
        v = (v > 0.f) ? v : 0.1f * v;
      }
      aA[tid * N2P + cc] = (_Float16)v;
    }
  }
  __syncthreads();

  // GEMM2: [16,256] @ Wd2[256,256]
  for (int nt = wave * 4; nt < wave * 4 + 4; ++nt) {
    v8f acc = {};
#pragma unroll
    for (int kt = 0; kt < 8; ++kt)
      acc = wmma16(load_A(aA, 0, kt * 32, N2P), load_Bfrag(Wd2F, kt * 16 + nt), acc);
    const int col = nt * 16 + (lane & 15);
    const int rb  = (lane < 16) ? 0 : 8;
    const float bias = (col < N2_) ? bd2[col] : 0.f;
#pragma unroll
    for (int r = 0; r < 8; ++r) hf[(rb + r) * N2P + col] = acc[r] + bias;
  }
  __syncthreads();

  if (tid < 16) {  // LN2 + leaky
    float m = 0.f, m2 = 0.f;
    for (int cc = 0; cc < N2_; ++cc) { const float v = hf[tid * N2P + cc]; m += v; m2 += v * v; }
    m *= (1.f / N2_); m2 = m2 * (1.f / N2_) - m * m;
    const float inv = rsqrtf(m2 + 1e-5f);
    for (int cc = 0; cc < N2P; ++cc) {
      float v = 0.f;
      if (cc < N2_) {
        v = (hf[tid * N2P + cc] - m) * inv * g2[cc] + be2[cc];
        v = (v > 0.f) ? v : 0.1f * v;
      }
      aA[tid * N2P + cc] = (_Float16)v;
    }
  }
  __syncthreads();

  // GEMM3: [16,256] @ Wd3[256,128] -> recon (cols < 116)
  for (int nt = wave * 2; nt < wave * 2 + 2; ++nt) {
    v8f acc = {};
#pragma unroll
    for (int kt = 0; kt < 8; ++kt)
      acc = wmma16(load_A(aA, 0, kt * 32, N2P), load_Bfrag(Wd3F, kt * 8 + nt), acc);
    const int col = nt * 16 + (lane & 15);
    const int rb  = (lane < 16) ? 0 : 8;
    if (col < N_) {
      const float bias = bd3[col];
#pragma unroll
      for (int r = 0; r < 8; ++r)
        recon[(size_t)(row0 + rb + r) * N_ + col] = acc[r] + bias;
    }
  }
}

// ----------------------------------------------------------- launch

extern "C" void kernel_launch(void* const* d_in, const int* in_sizes, int n_in,
                              void* d_out, int out_size, void* d_ws, size_t ws_size,
                              hipStream_t stream) {
  const float* x     = (const float*)d_in[0];
  const float* sc    = (const float*)d_in[1];
  const float* Wf    = (const float*)d_in[2];
  const float* bf    = (const float*)d_in[3];
  const float* gf    = (const float*)d_in[4];
  const float* betaf = (const float*)d_in[5];
  const float* We    = (const float*)d_in[6];
  const float* be    = (const float*)d_in[7];
  const float* omega = (const float*)d_in[8];
  const float* Wm    = (const float*)d_in[9];
  const float* bm    = (const float*)d_in[10];
  const float* w_snn = (const float*)d_in[11];
  const float* b_snn = (const float*)d_in[12];
  const float* Wd1   = (const float*)d_in[13];
  const float* bd1   = (const float*)d_in[14];
  const float* g1    = (const float*)d_in[15];
  const float* be1   = (const float*)d_in[16];
  const float* Wd2   = (const float*)d_in[17];
  const float* bd2   = (const float*)d_in[18];
  const float* g2    = (const float*)d_in[19];
  const float* be2   = (const float*)d_in[20];
  const float* Wd3   = (const float*)d_in[21];
  const float* bd3   = (const float*)d_in[22];

  char* ws = (char*)d_ws;
  size_t off = 0;
  auto carve = [&](size_t bytes) { char* p = ws + off; off += (bytes + 255) & ~(size_t)255; return p; };

  _Float16* WfF  = (_Float16*)carve((size_t)4 * 16 * 512 * 2);
  _Float16* WeF  = (_Float16*)carve((size_t)8 * 8  * 512 * 2);
  _Float16* WmF  = (_Float16*)carve((size_t)4 * 8  * 512 * 2);
  _Float16* Wd1F = (_Float16*)carve((size_t)4 * 16 * 512 * 2);
  _Float16* Wd2F = (_Float16*)carve((size_t)8 * 16 * 512 * 2);
  _Float16* Wd3F = (_Float16*)carve((size_t)8 * 8  * 512 * 2);

  const size_t BTN = (size_t)B_ * T_ * N_;
  float* gamma = (float*)carve(BTN * 4);
  float* mask  = (float*)carve(BTN * 4);
  float* u     = (float*)carve(BTN * 4);
  float* Icur  = (float*)carve(BTN * 4);
  float* snn   = (float*)carve(BTN * 4);

  float* recon  = (float*)d_out;
  float* spikes = (float*)d_out + BTN;

  // 1) pack weights into WMMA B-fragment layout
  pack_weight_frags<<<64, 256, 0, stream>>>(Wf,  WfF,  N_,  H_,  4, 16);
  pack_weight_frags<<<64, 256, 0, stream>>>(We,  WeF,  H_,  N_,  8, 8);
  pack_weight_frags<<<64, 256, 0, stream>>>(Wm,  WmF,  N_,  N_,  4, 8);
  pack_weight_frags<<<64, 256, 0, stream>>>(Wd1, Wd1F, N_,  N2_, 4, 16);
  pack_weight_frags<<<64, 256, 0, stream>>>(Wd2, Wd2F, N2_, N2_, 8, 16);
  pack_weight_frags<<<64, 256, 0, stream>>>(Wd3, Wd3F, N2_, N_,  8, 8);

  const int row_tiles = (B_ * T_) / 16;  // 4096

  // 2) fused encoder -> gamma
  encoder_kernel<<<row_tiles, 128, 0, stream>>>(x, WfF, bf, gf, betaf, WeF, be, gamma);

  // 3) Kuramoto scan (one workgroup per batch element) -> mask, u
  kuramoto_kernel<<<B_, 128, 0, stream>>>(sc, gamma, omega, w_snn, mask, u);

  // 4) gating GEMM -> I
  gating_kernel<<<row_tiles, 128, 0, stream>>>(mask, u, WmF, bm, b_snn, Icur);

  // 5) LIF recurrence -> snn state + spikes (output)
  lif_kernel<<<(B_ * N_ + 127) / 128, 128, 0, stream>>>(Icur, snn, spikes);

  // 6) fused 3-GEMM decoder -> recon (output)
  decoder_kernel<<<row_tiles, 128, 0, stream>>>(snn, Wd1F, bd1, g1, be1,
                                                Wd2F, bd2, g2, be2, Wd3F, bd3, recon);
}
